// GRUDecoder_81930796138843
// MI455X (gfx1250) — matrix operand order repaired and checked by reading, not compiled, not run on hardware
//
#include <hip/hip_runtime.h>
#include <hip/hip_bf16.h>

typedef __attribute__((ext_vector_type(16))) _Float16 v16h;
typedef __attribute__((ext_vector_type(8)))  float    v8f;

#define B_ 256
#define T_ 1000
#define H_ 64
#define D_ 128
#define G_ 192   // 3*H

// ---- A-operand (16x32 f16, M x K) gather per ISA layout:
// lane L: m = L&15, half = L>>4; elem e -> k = (e>>3)*16 + half*8 + (e&7)
__device__ __forceinline__ v16h lds_load_a16(const _Float16* src, int ldk, int lane) {
  const int m = lane & 15, hf = lane >> 4;
  const _Float16* p = src + m * ldk + hf * 8;
  v16h a;
#pragma unroll
  for (int e = 0; e < 8; ++e) a[e] = p[e];
#pragma unroll
  for (int e = 0; e < 8; ++e) a[8 + e] = p[16 + e];
  return a;
}

// ---- B-operand (32x16 f16, K x N), stored as W[n][k] row-major:
// lane L: n = L&15, elem e -> k = (e>>3)*16 + (L>>4)*8 + (e&7)
__device__ __forceinline__ v16h lds_load_b16(const _Float16* src, int ldk, int lane) {
  const int n = lane & 15, kh = lane >> 4;
  const _Float16* p = src + n * ldk + kh * 8;
  v16h b;
#pragma unroll
  for (int e = 0; e < 8; ++e) b[e] = p[e];
#pragma unroll
  for (int e = 0; e < 8; ++e) b[8 + e] = p[16 + e];
  return b;
}

// fast 1-ULP reciprocal (single v_rcp_f32) -- avoids the IEEE div expansion
__device__ __forceinline__ float fast_rcp(float x) { return __builtin_amdgcn_rcpf(x); }

__device__ __forceinline__ float sigmoid_f(float x) {
  return fast_rcp(1.0f + __expf(-x));                 // v_exp + v_rcp
}
__device__ __forceinline__ float tanh_f(float x) {
  x = fminf(fmaxf(x, -15.0f), 15.0f);
  const float e = __expf(2.0f * x);
  return (e - 1.0f) * fast_rcp(e + 1.0f);             // v_exp + v_rcp + fma
}

__global__ __launch_bounds__(128, 1)
void GRUDecoder_81930796138843_kernel(const float* __restrict__ hidden,
                                      const float* __restrict__ tf,
                                      const float* __restrict__ Wih,
                                      const float* __restrict__ Whh,
                                      const float* __restrict__ bih,
                                      const float* __restrict__ bhh,
                                      const float* __restrict__ fcw,
                                      const float* __restrict__ fcb,
                                      float* __restrict__ out) {
  extern __shared__ char smem[];
  _Float16* sWih = (_Float16*)smem;           // [192][128]
  _Float16* sWhh = sWih + G_ * D_;            // [192][64]
  _Float16* sFc  = sWhh + G_ * H_;            // [128][64]
  _Float16* sX   = sFc  + D_ * H_;            // [2][16][128] double-buffered x
  _Float16* sH16 = sX   + 2 * 16 * D_;        // [16][64]
  float* sH32 = (float*)(sH16 + 16 * H_);     // [16][64]
  float* sBih = sH32 + 16 * H_;               // [192]
  float* sBhh = sBih + G_;                    // [192]
  float* sFcb = sBhh + G_;                    // [128]

  const int tid  = threadIdx.x;
  const int lane = tid & 31;
  const int wave = tid >> 5;          // wave w owns hidden units [16w, 16w+16)
  const int w16  = wave * 16;
  const int b0   = blockIdx.x * 16;   // 16 batch rows per block, independent chains

  // ---- one-time staging: weights -> f16 LDS, h0, biases, x[0]=0
  for (int i = tid; i < G_ * D_;     i += 128) sWih[i] = (_Float16)Wih[i];
  for (int i = tid; i < G_ * H_;     i += 128) sWhh[i] = (_Float16)Whh[i];
  for (int i = tid; i < D_ * H_;     i += 128) sFc[i]  = (_Float16)fcw[i];
  for (int i = tid; i < 2 * 16 * D_; i += 128) sX[i]   = (_Float16)0.0f;
  for (int i = tid; i < 16 * H_;     i += 128) {
    const float h0 = hidden[(size_t)(b0 + (i >> 6)) * H_ + (i & 63)];
    sH32[i] = h0; sH16[i] = (_Float16)h0;
  }
  for (int i = tid; i < G_; i += 128) { sBih[i] = bih[i]; sBhh[i] = bhh[i]; }
  for (int i = tid; i < D_; i += 128) sFcb[i] = fcb[i];
  __syncthreads();

  const int n  = lane & 15;
  const int mh = (lane >> 4) * 8;

  // ---- hoist loop-invariant B operands into VGPRs (compiler can't prove LDS const)
  v16h Bhh[3][2];                     // W_hh tiles for r/z/n at this wave's columns
#pragma unroll
  for (int g = 0; g < 3; ++g)
#pragma unroll
    for (int kc = 0; kc < 2; ++kc)
      Bhh[g][kc] = lds_load_b16(sWhh + (g * 64 + w16) * H_ + kc * 32, H_, lane);
  v16h Bfc[2][2];                     // fc_w tiles: out cols [32w, 32w+32)
#pragma unroll
  for (int tt = 0; tt < 2; ++tt)
#pragma unroll
    for (int kc = 0; kc < 2; ++kc)
      Bfc[tt][kc] = lds_load_b16(sFc + (wave * 32 + tt * 16) * H_ + kc * 32, H_, lane);

  const float bR  = sBih[w16 + n]        + sBhh[w16 + n];
  const float bZ  = sBih[64 + w16 + n]   + sBhh[64 + w16 + n];
  const float bNi = sBih[128 + w16 + n];
  const float bNh = sBhh[128 + w16 + n];
  const float bF0 = sFcb[wave * 32 + n];
  const float bF1 = sFcb[wave * 32 + 16 + n];

  int buf = 0;
  const int prow = tid >> 3;          // x-staging: 8 threads per row
  const int pd0  = (tid & 7) * 16;

#pragma unroll 1
  for (int t = 0; t < T_; ++t) {
    __syncthreads();                  // h(t-1), x[buf] visible

    // ---- minimal read phase: only sH16/sH32 are written later this step
    v16h ah[2];
#pragma unroll
    for (int kc = 0; kc < 2; ++kc) ah[kc] = lds_load_a16(sH16 + kc * 32, H_, lane);
    float hp[8];
#pragma unroll
    for (int r = 0; r < 8; ++r) hp[r] = sH32[(mh + r) * H_ + w16 + n];

    __syncthreads();                  // all h reads done -> safe to overwrite below

    // prefetch x for step t+1 ( x[t+1] = tf[:, t, :] ), coalesced 64B/thread
    float pf[16];
    {
      const float* src = tf + ((size_t)(b0 + prow) * T_ + t) * D_ + pd0;
#pragma unroll
      for (int j = 0; j < 16; ++j) pf[j] = src[j];
      __builtin_prefetch(src + D_, 0, 1);
    }

    // ---- gi: x(16x128) @ W_ih^T for r/z/n tiles at this wave's columns (12 WMMA)
    v16h ax[4];
#pragma unroll
    for (int kc = 0; kc < 4; ++kc)
      ax[kc] = lds_load_a16(sX + buf * 16 * D_ + kc * 32, D_, lane);

    v8f accR = {}, accZ = {}, accNi = {};
#pragma unroll
    for (int kc = 0; kc < 4; ++kc) {
      v16h bm = lds_load_b16(sWih + (w16) * D_ + kc * 32, D_, lane);
      accR = __builtin_amdgcn_wmma_f32_16x16x32_f16(false, ax[kc], false, bm,
                                                    (short)0, accR, false, false);
      bm = lds_load_b16(sWih + (64 + w16) * D_ + kc * 32, D_, lane);
      accZ = __builtin_amdgcn_wmma_f32_16x16x32_f16(false, ax[kc], false, bm,
                                                    (short)0, accZ, false, false);
      bm = lds_load_b16(sWih + (128 + w16) * D_ + kc * 32, D_, lane);
      accNi = __builtin_amdgcn_wmma_f32_16x16x32_f16(false, ax[kc], false, bm,
                                                     (short)0, accNi, false, false);
    }

    // ---- FC projection of h(t-1) -> out[:, t-1, 32w:32w+32]  (4 WMMA, reuses ah)
    if (t > 0) {
#pragma unroll
      for (int tt = 0; tt < 2; ++tt) {
        v8f acc = {};
#pragma unroll
        for (int kc = 0; kc < 2; ++kc)
          acc = __builtin_amdgcn_wmma_f32_16x16x32_f16(false, ah[kc], false, Bfc[tt][kc],
                                                       (short)0, acc, false, false);
        const float bias = tt ? bF1 : bF0;
#pragma unroll
        for (int r = 0; r < 8; ++r)
          __builtin_nontemporal_store(acc[r] + bias,
              &out[((size_t)(b0 + mh + r) * T_ + (t - 1)) * D_ + wave * 32 + tt * 16 + n]);
      }
    }

    // ---- gh: h(t-1) @ W_hh^T (6 WMMA; r/z fused into gi accumulators)
#pragma unroll
    for (int kc = 0; kc < 2; ++kc)
      accR = __builtin_amdgcn_wmma_f32_16x16x32_f16(false, ah[kc], false, Bhh[0][kc],
                                                    (short)0, accR, false, false);
#pragma unroll
    for (int kc = 0; kc < 2; ++kc)
      accZ = __builtin_amdgcn_wmma_f32_16x16x32_f16(false, ah[kc], false, Bhh[1][kc],
                                                    (short)0, accZ, false, false);
    v8f accNh = {};
#pragma unroll
    for (int kc = 0; kc < 2; ++kc)
      accNh = __builtin_amdgcn_wmma_f32_16x16x32_f16(false, ah[kc], false, Bhh[2][kc],
                                                     (short)0, accNh, false, false);

    // ---- in-register GRU update for this wave's 16x16 block, write h(t)
#pragma unroll
    for (int r = 0; r < 8; ++r) {
      const float rr = sigmoid_f(accR[r] + bR);
      const float zz = sigmoid_f(accZ[r] + bZ);
      const float nn = tanh_f(fmaf(rr, accNh[r] + bNh, accNi[r] + bNi));
      const float hn = fmaf(zz, hp[r] - nn, nn);      // (1-z)*n + z*h
      const int idx = (mh + r) * H_ + w16 + n;
      sH32[idx] = hn;
      sH16[idx] = (_Float16)hn;
    }

    // stage prefetched x into the other buffer (consumed next step)
    {
      _Float16* dst = sX + (buf ^ 1) * 16 * D_ + prow * D_ + pd0;
#pragma unroll
      for (int j = 0; j < 16; ++j) dst[j] = (_Float16)pf[j];
    }
    buf ^= 1;
  }

  // ---- final output row t = T-1
  __syncthreads();
  {
    v16h ah[2];
#pragma unroll
    for (int kc = 0; kc < 2; ++kc) ah[kc] = lds_load_a16(sH16 + kc * 32, H_, lane);
#pragma unroll
    for (int tt = 0; tt < 2; ++tt) {
      v8f acc = {};
#pragma unroll
      for (int kc = 0; kc < 2; ++kc)
        acc = __builtin_amdgcn_wmma_f32_16x16x32_f16(false, ah[kc], false, Bfc[tt][kc],
                                                     (short)0, acc, false, false);
      const float bias = tt ? bF1 : bF0;
#pragma unroll
      for (int r = 0; r < 8; ++r)
        __builtin_nontemporal_store(acc[r] + bias,
            &out[((size_t)(b0 + mh + r) * T_ + (T_ - 1)) * D_ + wave * 32 + tt * 16 + n]);
    }
  }
}

extern "C" void kernel_launch(void* const* d_in, const int* in_sizes, int n_in,
                              void* d_out, int out_size, void* d_ws, size_t ws_size,
                              hipStream_t stream) {
  (void)in_sizes; (void)n_in; (void)d_ws; (void)ws_size; (void)out_size;
  const float* hidden = (const float*)d_in[0];
  // d_in[1] = target_length (compile-time constant T_ = 1000)
  const float* tf  = (const float*)d_in[2];
  const float* Wih = (const float*)d_in[3];
  const float* Whh = (const float*)d_in[4];
  const float* bih = (const float*)d_in[5];
  const float* bhh = (const float*)d_in[6];
  const float* fcw = (const float*)d_in[7];
  const float* fcb = (const float*)d_in[8];
  float* out = (float*)d_out;

  // f16 LDS: (192*128 + 192*64 + 128*64 + 2*16*128 + 16*64) * 2 = 100352 B
  // f32 LDS: (16*64 + 2*192 + 128) * 4                          =   6144 B
  const size_t lds_bytes = 106496;   // well under the 320 KB WGP limit

  GRUDecoder_81930796138843_kernel<<<dim3(B_ / 16), dim3(128), lds_bytes, stream>>>(
      hidden, tf, Wih, Whh, bih, bhh, fcw, fcb, out);
}
